// stft_43284680409346
// MI455X (gfx1250) — matrix operand order repaired and checked by reading, not compiled, not run on hardware
//
#include <hip/hip_runtime.h>
#include <stdint.h>

typedef _Float16 v16h __attribute__((ext_vector_type(16)));
typedef _Float16 v8h  __attribute__((ext_vector_type(8)));
typedef _Float16 v4h  __attribute__((ext_vector_type(4)));
typedef float    v8f  __attribute__((ext_vector_type(8)));
typedef float    v4f  __attribute__((ext_vector_type(4)));
typedef int      v4i_t __attribute__((ext_vector_type(4)));

#define N_FFT_C   1024
#define N_HOP_C   512
#define NB_FILT   513
#define N_FRAMES  2047
#define T_LEN     1048576

#define BM 64
#define BN 128
#define BK 64
#define LDS_STR (BK + 8)          // halves; 144B row stride -> conflict-free b128 reads

// per-buffer LDS layout (halves)
#define A_PLANE (BM * LDS_STR)    // 4608
#define B_PLANE (BN * LDS_STR)    // 9216
#define BUF_H   (2 * A_PLANE + 2 * B_PLANE)   // 27648 halves = 55296 B

#if defined(__HIP_DEVICE_COMPILE__) && __has_builtin(__builtin_amdgcn_global_load_async_to_lds_b128)
#define HAVE_ASYNC_LDS 1
#else
#define HAVE_ASYNC_LDS 0
#endif

__device__ __forceinline__ void cvt_split(float x, _Float16& h, _Float16& l) {
    h = (_Float16)x;
    l = (_Float16)(x - (float)h);
}

__device__ __forceinline__ v16h cat8(v8h a, v8h b) {
    return __builtin_shufflevector(a, b, 0,1,2,3,4,5,6,7,8,9,10,11,12,13,14,15);
}

__device__ __forceinline__ void wait_async0() {
#if __has_builtin(__builtin_amdgcn_s_wait_asynccnt)
    __builtin_amdgcn_s_wait_asynccnt(0);
#else
    asm volatile("s_wait_asynccnt 0x0" ::: "memory");
#endif
}

#if HAVE_ASYNC_LDS
typedef __attribute__((address_space(1))) v4i_t gv4i;
typedef __attribute__((address_space(3))) v4i_t lv4i;
__device__ __forceinline__ void async_cp16(const _Float16* gp, _Float16* lp) {
    __builtin_amdgcn_global_load_async_to_lds_b128((gv4i*)gp, (lv4i*)lp, 0, 0);
}
#endif

// ---------------- pass 1: f32 -> f16 hi/lo planes ----------------
__global__ __launch_bounds__(256) void split_kernel(
    const float* __restrict__ src, _Float16* __restrict__ hi,
    _Float16* __restrict__ lo, int n4)
{
    int i = blockIdx.x * 256 + threadIdx.x;
    if (i < n4) {
        v4f v = ((const v4f*)src)[i];
        v4h h, l;
        #pragma unroll
        for (int j = 0; j < 4; ++j) {
            _Float16 hh, ll;
            cvt_split(v[j], hh, ll);
            h[j] = hh;
            l[j] = ll;
        }
        ((v4h*)hi)[i] = h;
        ((v4h*)lo)[i] = l;
    }
}

// ---------------- pass 2: WMMA GEMM from pre-split planes ----------------
__device__ __forceinline__ void stage_tiles(
    _Float16* __restrict__ buf,
    const _Float16* __restrict__ kh, const _Float16* __restrict__ kl,
    const _Float16* __restrict__ sh, const _Float16* __restrict__ sl,
    int mtile, int ntile, int k0, int tid)
{
    _Float16* Ah = buf;
    _Float16* Al = buf + A_PLANE;
    _Float16* Bh = buf + 2 * A_PLANE;
    _Float16* Bl = buf + 2 * A_PLANE + B_PLANE;

    // A: 64 rows x 64 halves x 2 planes = 1024 x 16B chunks -> 4/thread
    #pragma unroll
    for (int i = 0; i < 4; ++i) {
        const int cid   = tid + 256 * i;
        const int plane = cid >> 9;
        const int row   = (cid >> 3) & 63;
        const int seg   = (cid & 7) * 8;
        int kr = mtile * BM + row;  kr = kr < NB_FILT ? kr : (NB_FILT - 1);
        const _Float16* gp = (plane ? kl : kh) + (size_t)kr * N_FFT_C + k0 + seg;
        _Float16*       lp = (plane ? Al : Ah) + row * LDS_STR + seg;
#if HAVE_ASYNC_LDS
        async_cp16(gp, lp);
#else
        *(v8h*)lp = *(const v8h*)gp;
#endif
    }
    // B: 128 rows x 64 halves x 2 planes = 2048 x 16B chunks -> 8/thread
    #pragma unroll
    for (int i = 0; i < 8; ++i) {
        const int cid   = tid + 256 * i;
        const int plane = cid >> 10;
        const int row   = (cid >> 3) & 127;
        const int seg   = (cid & 7) * 8;
        int fg = ntile * BN + row;  fg = fg < N_FRAMES ? fg : (N_FRAMES - 1);
        const _Float16* gp = (plane ? sl : sh) + (size_t)fg * N_HOP_C + k0 + seg;
        _Float16*       lp = (plane ? Bl : Bh) + row * LDS_STR + seg;
#if HAVE_ASYNC_LDS
        async_cp16(gp, lp);
#else
        *(v8h*)lp = *(const v8h*)gp;
#endif
    }
}

__global__ __launch_bounds__(256) void stft_wmma_v2(
    const _Float16* __restrict__ smp_hi, const _Float16* __restrict__ smp_lo,
    const _Float16* __restrict__ rk_hi,  const _Float16* __restrict__ rk_lo,
    const _Float16* __restrict__ ik_hi,  const _Float16* __restrict__ ik_lo,
    float* __restrict__ out)
{
    __shared__ __align__(16) _Float16 ldsbuf[2][BUF_H];   // ~108 KB: 2 blocks/WGP

    const int ntile = blockIdx.x;
    const int mtile = blockIdx.y;
    const int bz    = blockIdx.z;
    const int batch = bz & 15;
    const int which = bz >> 4;

    const _Float16* kh = which ? ik_hi : rk_hi;
    const _Float16* kl = which ? ik_lo : rk_lo;
    const _Float16* sh = smp_hi + (size_t)batch * T_LEN;
    const _Float16* sl = smp_lo + (size_t)batch * T_LEN;
    float* __restrict__ obase = out
        + (size_t)which * ((size_t)16 * NB_FILT * N_FRAMES)
        + (size_t)batch * ((size_t)NB_FILT * N_FRAMES);

    const int tid    = threadIdx.x;
    const int lane   = tid & 31;
    const int wid    = tid >> 5;
    const int waveM  = wid & 3;
    const int waveN  = wid >> 2;
    const int laneHi = lane >> 4;
    const int lm     = lane & 15;

    v8f acc[4] = {};

    // prologue: stage chunk 0 into buffer 0
    stage_tiles(ldsbuf[0], kh, kl, sh, sl, mtile, ntile, 0, tid);
#if HAVE_ASYNC_LDS
    wait_async0();
#endif
    __syncthreads();

    for (int k0 = 0; k0 < N_FFT_C; k0 += BK) {
        const int cur = (k0 / BK) & 1;
        // issue async copies for next chunk into the other buffer
        if (k0 + BK < N_FFT_C)
            stage_tiles(ldsbuf[cur ^ 1], kh, kl, sh, sl, mtile, ntile, k0 + BK, tid);

        const _Float16* Ah = ldsbuf[cur];
        const _Float16* Al = Ah + A_PLANE;
        const _Float16* Bh = Ah + 2 * A_PLANE;
        const _Float16* Bl = Ah + 2 * A_PLANE + B_PLANE;

        #pragma unroll
        for (int ks = 0; ks < BK; ks += 32) {
            const int aoff = (waveM * 16 + lm) * LDS_STR + ks + laneHi * 8;
            v16h ah = cat8(*(const v8h*)(Ah + aoff), *(const v8h*)(Ah + aoff + 16));
            v16h al = cat8(*(const v8h*)(Al + aoff), *(const v8h*)(Al + aoff + 16));

            #pragma unroll
            for (int s = 0; s < 4; ++s) {
                const int boff = (waveN * 64 + s * 16 + lm) * LDS_STR + ks + laneHi * 8;
                v16h bh = cat8(*(const v8h*)(Bh + boff), *(const v8h*)(Bh + boff + 16));
                v16h bl = cat8(*(const v8h*)(Bl + boff), *(const v8h*)(Bl + boff + 16));
                acc[s] = __builtin_amdgcn_wmma_f32_16x16x32_f16(false, ah, false, bh,
                                                                (short)0, acc[s], false, false);
                acc[s] = __builtin_amdgcn_wmma_f32_16x16x32_f16(false, ah, false, bl,
                                                                (short)0, acc[s], false, false);
                acc[s] = __builtin_amdgcn_wmma_f32_16x16x32_f16(false, al, false, bh,
                                                                (short)0, acc[s], false, false);
            }
        }
#if HAVE_ASYNC_LDS
        wait_async0();                 // my next-chunk copies have landed
#endif
        __syncthreads();               // everyone done reading cur + next chunk visible
    }

    const int colbase = ntile * BN + waveN * 64 + lm;
    const int rowbase = mtile * BM + waveM * 16 + laneHi * 8;
    #pragma unroll
    for (int s = 0; s < 4; ++s) {
        const int col = colbase + s * 16;
        if (col < N_FRAMES) {
            #pragma unroll
            for (int j = 0; j < 8; ++j) {
                const int row = rowbase + j;
                if (row < NB_FILT)
                    obase[(size_t)row * N_FRAMES + col] = acc[s][j];
            }
        }
    }
}

// ---------------- fallback: round-1 fused kernel (used if d_ws too small) ----
__global__ __launch_bounds__(256) void stft_wmma_v1(
    const float* __restrict__ sample, const float* __restrict__ rker,
    const float* __restrict__ iker, float* __restrict__ out)
{
    __shared__ __align__(16) _Float16 Ahi[A_PLANE];
    __shared__ __align__(16) _Float16 Alo[A_PLANE];
    __shared__ __align__(16) _Float16 Bhi[B_PLANE];
    __shared__ __align__(16) _Float16 Blo[B_PLANE];

    const int ntile = blockIdx.x, mtile = blockIdx.y, bz = blockIdx.z;
    const int batch = bz & 15, which = bz >> 4;
    const float* __restrict__ ker = which ? iker : rker;
    const float* __restrict__ smp = sample + (size_t)batch * T_LEN;
    float* __restrict__ obase = out
        + (size_t)which * ((size_t)16 * NB_FILT * N_FRAMES)
        + (size_t)batch * ((size_t)NB_FILT * N_FRAMES);

    const int tid = threadIdx.x, lane = tid & 31, wid = tid >> 5;
    const int waveM = wid & 3, waveN = wid >> 2;
    const int laneHi = lane >> 4, lm = lane & 15;

    v8f acc[4] = {};

    for (int k0 = 0; k0 < N_FFT_C; k0 += BK) {
        {
            const int r = tid >> 2, cseg = (tid & 3) * 16;
            int kr = mtile * BM + r; kr = kr < NB_FILT ? kr : (NB_FILT - 1);
            const float* gp = ker + (size_t)kr * N_FFT_C + k0 + cseg;
            _Float16* lph = Ahi + r * LDS_STR + cseg;
            _Float16* lpl = Alo + r * LDS_STR + cseg;
            #pragma unroll
            for (int i = 0; i < 4; ++i) {
                v4f v = *(const v4f*)(gp + i * 4);
                #pragma unroll
                for (int j = 0; j < 4; ++j) cvt_split(v[j], lph[i*4+j], lpl[i*4+j]);
            }
        }
        {
            const int f = tid >> 1, seg = (tid & 1) * 32;
            int fg = ntile * BN + f; fg = fg < N_FRAMES ? fg : (N_FRAMES - 1);
            const float* gp = smp + (size_t)fg * N_HOP_C + k0 + seg;
            _Float16* lph = Bhi + f * LDS_STR + seg;
            _Float16* lpl = Blo + f * LDS_STR + seg;
            #pragma unroll
            for (int i = 0; i < 8; ++i) {
                v4f v = *(const v4f*)(gp + i * 4);
                #pragma unroll
                for (int j = 0; j < 4; ++j) cvt_split(v[j], lph[i*4+j], lpl[i*4+j]);
            }
        }
        __syncthreads();
        #pragma unroll
        for (int ks = 0; ks < BK; ks += 32) {
            const int aoff = (waveM * 16 + lm) * LDS_STR + ks + laneHi * 8;
            v16h ah = cat8(*(const v8h*)(Ahi + aoff), *(const v8h*)(Ahi + aoff + 16));
            v16h al = cat8(*(const v8h*)(Alo + aoff), *(const v8h*)(Alo + aoff + 16));
            #pragma unroll
            for (int s = 0; s < 4; ++s) {
                const int boff = (waveN * 64 + s * 16 + lm) * LDS_STR + ks + laneHi * 8;
                v16h bh = cat8(*(const v8h*)(Bhi + boff), *(const v8h*)(Bhi + boff + 16));
                v16h bl = cat8(*(const v8h*)(Blo + boff), *(const v8h*)(Blo + boff + 16));
                acc[s] = __builtin_amdgcn_wmma_f32_16x16x32_f16(false, ah, false, bh,
                                                                (short)0, acc[s], false, false);
                acc[s] = __builtin_amdgcn_wmma_f32_16x16x32_f16(false, ah, false, bl,
                                                                (short)0, acc[s], false, false);
                acc[s] = __builtin_amdgcn_wmma_f32_16x16x32_f16(false, al, false, bh,
                                                                (short)0, acc[s], false, false);
            }
        }
        __syncthreads();
    }

    const int colbase = ntile * BN + waveN * 64 + lm;
    const int rowbase = mtile * BM + waveM * 16 + laneHi * 8;
    #pragma unroll
    for (int s = 0; s < 4; ++s) {
        const int col = colbase + s * 16;
        if (col < N_FRAMES) {
            #pragma unroll
            for (int j = 0; j < 8; ++j) {
                const int row = rowbase + j;
                if (row < NB_FILT)
                    obase[(size_t)row * N_FRAMES + col] = acc[s][j];
            }
        }
    }
}

extern "C" void kernel_launch(void* const* d_in, const int* in_sizes, int n_in,
                              void* d_out, int out_size, void* d_ws, size_t ws_size,
                              hipStream_t stream) {
    (void)in_sizes; (void)n_in; (void)out_size;
    const float* sample = (const float*)d_in[0];
    const float* rker   = (const float*)d_in[1];
    const float* iker   = (const float*)d_in[2];
    float* out          = (float*)d_out;

    const size_t nS = (size_t)16 * T_LEN;            // 16,777,216
    const size_t nK = (size_t)NB_FILT * N_FFT_C;     // 525,312
    const size_t need = (2 * nS + 4 * nK) * 2;       // bytes of f16 planes

    dim3 grid(16, 9, 32);

    if (ws_size >= need) {
        _Float16* ws = (_Float16*)d_ws;
        _Float16* sh = ws;
        _Float16* sl = sh + nS;
        _Float16* rh = sl + nS;
        _Float16* rl = rh + nK;
        _Float16* ih = rl + nK;
        _Float16* il = ih + nK;

        split_kernel<<<dim3((unsigned)(nS / 4 / 256)), dim3(256), 0, stream>>>(
            sample, sh, sl, (int)(nS / 4));
        split_kernel<<<dim3((unsigned)((nK / 4 + 255) / 256)), dim3(256), 0, stream>>>(
            rker, rh, rl, (int)(nK / 4));
        split_kernel<<<dim3((unsigned)((nK / 4 + 255) / 256)), dim3(256), 0, stream>>>(
            iker, ih, il, (int)(nK / 4));
        stft_wmma_v2<<<grid, dim3(256), 0, stream>>>(sh, sl, rh, rl, ih, il, out);
    } else {
        stft_wmma_v1<<<grid, dim3(256), 0, stream>>>(sample, rker, iker, out);
    }
}
